// OumnixSimpleAI_42331197669523
// MI455X (gfx1250) — matrix operand order repaired and checked by reading, not compiled, not run on hardware
//
#include <hip/hip_runtime.h>
#include <hip/hip_bf16.h>
#include <cstdint>

// ---------------------------------------------------------------------------
// Sliding-window + global attention block for MI455X (gfx1250, wave32, WMMA)
// GEMMs: v_wmma_f32_16x16x32_f16, A staged via global_load_async_to_lds_b128
// (double-buffered, ASYNCcnt), B read directly from pre-transposed weights.
// ---------------------------------------------------------------------------

typedef _Float16 f16;
typedef __attribute__((ext_vector_type(16))) _Float16 v16h;
typedef __attribute__((ext_vector_type(8)))  float    v8f;

#define DIM   1024
#define NSEQ  2048
#define BATCH 2
#define HEADS 16
#define HD    64
#define WIN   64
#define BN    (BATCH * NSEQ)

union HV { uint4 u[2]; v16h v; };

// --- CDNA5 WMMA fragment layout (16-bit data, wave32) -----------------------
// A (16x32): lane L holds row M=L&15; element e -> K = ka + (e<8 ? e : 8+e)
//            i.e. two contiguous 8-element runs at [ka, ka+8) and [16+ka, +8),
//            ka = 8*(L>=16).
// B (32x16): lane L holds col N=L&15; element e -> K = kb/2 + (e<8 ? e : 8+e),
//            same runs as A, kb = 16*(L>=16).
__device__ __forceinline__ v8f wmma16(v16h a, v16h b, v8f c) {
    return __builtin_amdgcn_wmma_f32_16x16x32_f16(
        false, a, false, b, (short)0, c, false, false);
}

__device__ __forceinline__ v16h load_frag_a(const f16* row, int ka) {
    HV h;
    h.u[0] = *(const uint4*)(row + ka);
    h.u[1] = *(const uint4*)(row + 16 + ka);
    return h.v;
}
__device__ __forceinline__ v16h load_frag_b(const f16* row, int kb) {
    HV h;
    h.u[0] = *(const uint4*)(row + (kb >> 1));
    h.u[1] = *(const uint4*)(row + 16 + (kb >> 1));
    return h.v;
}

// --- gfx1250 async global->LDS copy (ASYNCcnt) ------------------------------
__device__ __forceinline__ void async_load_b128(unsigned lds_byte_addr, const void* gaddr) {
    asm volatile("global_load_async_to_lds_b128 %0, %1, off"
                 :: "v"(lds_byte_addr), "v"((unsigned long long)gaddr)
                 : "memory");
}
__device__ __forceinline__ void wait_async0() {
    asm volatile("s_wait_asynccnt 0x0" ::: "memory");
}

// ---------------------------------------------------------------------------
// Kernel 1a: fp32 -> fp16 conversion (x)
// ---------------------------------------------------------------------------
__global__ void cvt_f32_f16(const float* __restrict__ in, f16* __restrict__ out, int n) {
    int i = blockIdx.x * blockDim.x + threadIdx.x;
    if (i < n) out[i] = (f16)in[i];
}

// ---------------------------------------------------------------------------
// Kernel 1b: fp32 -> fp16 + transpose (weights, DIM x DIM): out[n][k]=in[k][n]
// ---------------------------------------------------------------------------
__global__ __launch_bounds__(256) void cvt_transpose_f16(
        const float* __restrict__ in, f16* __restrict__ out) {
    __shared__ f16 tile[32][33];
    const int bx = blockIdx.x * 32, by = blockIdx.y * 32;
    const int tx = threadIdx.x & 31, ty = threadIdx.x >> 5;  // ty 0..7
#pragma unroll
    for (int i = 0; i < 32; i += 8)
        tile[ty + i][tx] = (f16)in[(size_t)(by + ty + i) * DIM + bx + tx];
    __syncthreads();
#pragma unroll
    for (int i = 0; i < 32; i += 8)
        out[(size_t)(bx + ty + i) * DIM + by + tx] = tile[tx][ty + i];
}

// ---------------------------------------------------------------------------
// Kernel 2: WMMA GEMM  C[M,N] = A[M,K] * Wt[N,K]^T   (A,Wt f16 row-major)
// Block: 256 threads (8 waves). Tile 32(M) x 128(N). K step 32.
// A double-buffered in LDS via async DMA; B fragments straight from global.
// ---------------------------------------------------------------------------
template <bool OUT_F32>
__global__ __launch_bounds__(256) void gemm_f16_wmma(
        const f16* __restrict__ A, const f16* __restrict__ Wt, void* __restrict__ Cout,
        int M, int N, int K) {
    __shared__ f16 As[2][32][32];
    const int t = threadIdx.x;
    const int lane = t & 31;
    const int w = t >> 5;                 // wave id 0..7 -> N sub-tile
    const int m0 = blockIdx.y * 32;
    const int n0 = blockIdx.x * 128;

    const unsigned lds0 = (unsigned)(unsigned long long)(void*)&As[0][0][0];
    const int ka = (lane & 16) >> 1;      // A-frag K base
    const int kb = lane & 16;             // B-frag K base
    const int m = lane & 15;
    const int ncol = n0 + (w << 4) + (lane & 15);
    const f16* wrow = Wt + (size_t)ncol * K;

    v8f c0 = {};
    v8f c1 = {};

    // stage first A tile (threads 0..127: 32 rows x 32 cols, 16B each)
    if (t < 128) {
        int row = t >> 2, seg = t & 3;
        async_load_b128(lds0 + (unsigned)(row * 64 + seg * 16),
                        A + (size_t)(m0 + row) * K + seg * 8);
    }

    int buf = 0;
    for (int k0 = 0; k0 < K; k0 += 32) {
        wait_async0();
        __syncthreads();                  // As[buf] ready, As[buf^1] free
        if (k0 + 32 < K) {
            if (t < 128) {
                int row = t >> 2, seg = t & 3;
                async_load_b128(lds0 + (unsigned)((buf ^ 1) * 2048 + row * 64 + seg * 16),
                                A + (size_t)(m0 + row) * K + k0 + 32 + seg * 8);
            }
            __builtin_prefetch(wrow + k0 + 64, 0, 1);  // global_prefetch_b8
        }
        v16h bfrag = load_frag_b(wrow + k0, kb);
        v16h a0 = load_frag_a(&As[buf][m][0], ka);
        v16h a1 = load_frag_a(&As[buf][16 + m][0], ka);
        c0 = wmma16(a0, bfrag, c0);
        c1 = wmma16(a1, bfrag, c1);
        buf ^= 1;
    }

    const int rbase = m0 + ((lane & 16) >> 1);
    if (OUT_F32) {
        float* C = (float*)Cout;
#pragma unroll
        for (int r = 0; r < 8; ++r) {
            C[(size_t)(rbase + r) * N + ncol]      = c0[r];
            C[(size_t)(rbase + 16 + r) * N + ncol] = c1[r];
        }
    } else {
        f16* C = (f16*)Cout;
#pragma unroll
        for (int r = 0; r < 8; ++r) {
            C[(size_t)(rbase + r) * N + ncol]      = (f16)c0[r];
            C[(size_t)(rbase + 16 + r) * N + ncol] = (f16)c1[r];
        }
    }
}

// ---------------------------------------------------------------------------
// Kernel 3: RoPE on q and k (in place). One thread per (token, head, pair).
// ---------------------------------------------------------------------------
__global__ void rope_qk(f16* __restrict__ q, f16* __restrict__ k) {
    int idx = blockIdx.x * blockDim.x + threadIdx.x;   // BN*HEADS*32 threads
    int p   = idx & 31;
    int h   = (idx >> 5) & (HEADS - 1);
    int tok = idx >> 9;                                // 0..BN-1
    if (tok >= BN) return;
    int pos = tok & (NSEQ - 1);                        // position in sequence
    float inv = __powf(10000.0f, -(float)p * (1.0f / 32.0f));
    float ang = (float)pos * inv;
    float c, s;
    __sincosf(ang, &c, &s);
    size_t base = (size_t)tok * DIM + h * HD;
    float q0 = (float)q[base + p], q1 = (float)q[base + p + 32];
    q[base + p]      = (f16)(q0 * c - q1 * s);
    q[base + p + 32] = (f16)(q1 * c + q0 * s);
    float k0 = (float)k[base + p], k1 = (float)k[base + p + 32];
    k[base + p]      = (f16)(k0 * c - k1 * s);
    k[base + p + 32] = (f16)(k1 * c + k0 * s);
}

// ---------------------------------------------------------------------------
// Kernel 4: attention. One wave handles 16 queries of one (b,h).
//   local: 96 keys [i0-64, i0+32) (band+causal masked), global: 32 keys ::64.
//   scores via WMMA, fp32 softmax in LDS, attn*V via WMMA.
// ---------------------------------------------------------------------------
__global__ __launch_bounds__(32) void attn_window_global(
        const f16* __restrict__ q, const f16* __restrict__ k,
        const f16* __restrict__ v, f16* __restrict__ ctx) {
    __shared__ float scoresL[16][96];
    __shared__ float scoresG[16][32];
    __shared__ f16   attnL[16][96];
    __shared__ f16   attnG[16][32];
    __shared__ f16   vT[64][96];    // vT[dim][key]
    __shared__ f16   vgT[64][32];

    const int lane = threadIdx.x;
    const int i0 = blockIdx.x * 16;
    const int h  = blockIdx.y;
    const int b  = blockIdx.z;
    const size_t boff = (size_t)b * NSEQ * DIM;
    const f16* qb = q + boff;
    const f16* kb = k + boff;
    const f16* vb = v + boff;
    f16* cb = ctx + boff;

    const float scale = 0.125f;                       // hd^-0.5
    const float slope = exp2f(-(float)(h + 1) * 0.5f);
    const int jbase = i0 - WIN;
    const int ka = (lane & 16) >> 1;
    const int kbo = lane & 16;
    const int m = lane & 15;

    // ---- stage V tiles transposed into LDS ----
    for (int c = 0; c < 96; ++c) {
        int j = jbase + c;
        int jc = j < 0 ? 0 : (j >= NSEQ ? NSEQ - 1 : j);
        vT[lane][c]      = vb[(size_t)jc * DIM + h * HD + lane];
        vT[lane + 32][c] = vb[(size_t)jc * DIM + h * HD + lane + 32];
    }
    for (int c = 0; c < 32; ++c) {
        int j = c * WIN;
        vgT[lane][c]      = vb[(size_t)j * DIM + h * HD + lane];
        vgT[lane + 32][c] = vb[(size_t)j * DIM + h * HD + lane + 32];
    }

    // ---- Q fragments (two K-steps of 32 over hd=64) ----
    const f16* qrow = qb + (size_t)(i0 + m) * DIM + h * HD;
    v16h aq0 = load_frag_a(qrow, ka);
    v16h aq1 = load_frag_a(qrow + 32, ka);

    // ---- local score tiles: 6 x (16 queries x 16 keys) ----
#pragma unroll
    for (int t = 0; t < 6; ++t) {
        int j = jbase + t * 16 + m;
        int jc = j < 0 ? 0 : (j >= NSEQ ? NSEQ - 1 : j);
        const f16* krow = kb + (size_t)jc * DIM + h * HD;
        v8f s = {};
        s = wmma16(aq0, load_frag_b(krow, kbo), s);
        s = wmma16(aq1, load_frag_b(krow + 32, kbo), s);
#pragma unroll
        for (int r = 0; r < 8; ++r)
            scoresL[r + ka][t * 16 + m] = s[r];
    }

    // ---- global score tiles: 2 x 16 keys (tokens 0,64,...,1984) ----
#pragma unroll
    for (int t = 0; t < 2; ++t) {
        int j = (t * 16 + m) * WIN;
        const f16* krow = kb + (size_t)j * DIM + h * HD;
        v8f s = {};
        s = wmma16(aq0, load_frag_b(krow, kbo), s);
        s = wmma16(aq1, load_frag_b(krow + 32, kbo), s);
#pragma unroll
        for (int r = 0; r < 8; ++r)
            scoresG[r + ka][t * 16 + m] = s[r];
    }
    __syncthreads();

    // ---- softmax: lanes 0-15 local rows, lanes 16-31 global rows ----
    if (lane < 16) {
        const int i = i0 + lane;
        float mx = -1e30f;
        for (int c = 0; c < 96; ++c) {
            int j = jbase + c;
            bool ok = (j >= 0) && (j <= i) && ((i - j) <= WIN);
            float sv = ok ? scoresL[lane][c] * scale + (float)(j - i) * slope : -1e30f;
            scoresL[lane][c] = sv;
            mx = fmaxf(mx, sv);
        }
        float sum = 0.0f;
        for (int c = 0; c < 96; ++c) {
            float e = __expf(scoresL[lane][c] - mx);
            sum += e;
            attnL[lane][c] = (f16)e;
        }
        float invs = 1.0f / sum;
        for (int c = 0; c < 96; ++c)
            attnL[lane][c] = (f16)((float)attnL[lane][c] * invs);
    } else {
        const int mr = lane - 16;
        float mx = -1e30f;
        for (int c = 0; c < 32; ++c) {
            float sv = scoresG[mr][c] * scale;
            scoresG[mr][c] = sv;
            mx = fmaxf(mx, sv);
        }
        float sum = 0.0f;
        for (int c = 0; c < 32; ++c) {
            float e = __expf(scoresG[mr][c] - mx);
            sum += e;
            attnG[mr][c] = (f16)e;
        }
        float invs = 1.0f / sum;
        for (int c = 0; c < 32; ++c)
            attnG[mr][c] = (f16)((float)attnG[mr][c] * invs);
    }
    __syncthreads();

    // ---- output: ctx(16 x 64) = attnL(16x96)*V + attnG(16x32)*Vg ----
#pragma unroll
    for (int nt = 0; nt < 4; ++nt) {
        v8f acc = {};
        const int n = (nt << 4) + m;
#pragma unroll
        for (int kc = 0; kc < 3; ++kc) {          // 96 local keys = 3 K-steps
            v16h a  = load_frag_a(&attnL[m][kc * 32], ka);
            v16h bv = load_frag_b(&vT[n][kc * 32], kbo);
            acc = wmma16(a, bv, acc);
        }
        {                                          // 32 global keys = 1 K-step
            v16h a  = load_frag_a(&attnG[m][0], ka);
            v16h bv = load_frag_b(&vgT[n][0], kbo);
            acc = wmma16(a, bv, acc);
        }
        const int rbase = (lane & 16) >> 1;
#pragma unroll
        for (int r = 0; r < 8; ++r)
            cb[(size_t)(i0 + rbase + r) * DIM + h * HD + n] = (f16)acc[r];
    }
}

// ---------------------------------------------------------------------------
// Host-side orchestration
// ---------------------------------------------------------------------------
extern "C" void kernel_launch(void* const* d_in, const int* in_sizes, int n_in,
                              void* d_out, int out_size, void* d_ws, size_t ws_size,
                              hipStream_t stream) {
    (void)in_sizes; (void)n_in; (void)out_size; (void)ws_size;
    const float* x  = (const float*)d_in[0];
    const float* Wq = (const float*)d_in[1];
    const float* Wk = (const float*)d_in[2];
    const float* Wv = (const float*)d_in[3];
    const float* Wo = (const float*)d_in[4];

    f16* p   = (f16*)d_ws;
    f16* xh  = p; p += (size_t)BN * DIM;
    f16* wqt = p; p += (size_t)DIM * DIM;   // transposed f16 weights: [n][k]
    f16* wkt = p; p += (size_t)DIM * DIM;
    f16* wvt = p; p += (size_t)DIM * DIM;
    f16* wot = p; p += (size_t)DIM * DIM;
    f16* qh  = p; p += (size_t)BN * DIM;
    f16* kh  = p; p += (size_t)BN * DIM;
    f16* vh  = p; p += (size_t)BN * DIM;
    f16* ctx = p; p += (size_t)BN * DIM;

    const int nx = BN * DIM;
    cvt_f32_f16<<<(nx + 255) / 256, 256, 0, stream>>>(x, xh, nx);
    dim3 tgrid(DIM / 32, DIM / 32);
    cvt_transpose_f16<<<tgrid, 256, 0, stream>>>(Wq, wqt);
    cvt_transpose_f16<<<tgrid, 256, 0, stream>>>(Wk, wkt);
    cvt_transpose_f16<<<tgrid, 256, 0, stream>>>(Wv, wvt);
    cvt_transpose_f16<<<tgrid, 256, 0, stream>>>(Wo, wot);

    dim3 ggrid(DIM / 128, BN / 32);  // (8, 128)
    gemm_f16_wmma<false><<<ggrid, 256, 0, stream>>>(xh, wqt, qh, BN, DIM, DIM);
    gemm_f16_wmma<false><<<ggrid, 256, 0, stream>>>(xh, wkt, kh, BN, DIM, DIM);
    gemm_f16_wmma<false><<<ggrid, 256, 0, stream>>>(xh, wvt, vh, BN, DIM, DIM);

    const int nrope = BN * HEADS * 32;
    rope_qk<<<(nrope + 255) / 256, 256, 0, stream>>>(qh, kh);

    dim3 agrid(NSEQ / 16, HEADS, BATCH);  // (128, 16, 2), 1 wave each
    attn_window_global<<<agrid, 32, 0, stream>>>(qh, kh, vh, ctx);

    gemm_f16_wmma<true><<<ggrid, 256, 0, stream>>>(ctx, wot, d_out, BN, DIM, DIM);
}
